// NodeUpdate_72516227826281
// MI455X (gfx1250) — matrix operand order repaired and checked by reading, not compiled, not run on hardware
//
#include <hip/hip_runtime.h>
#include <hip/hip_bf16.h>
#include <math.h>

// ---------------------------------------------------------------------------
// Types for WMMA fragments (wave32, CDNA5)
// ---------------------------------------------------------------------------
typedef __attribute__((ext_vector_type(16))) _Float16 v16h;
typedef __attribute__((ext_vector_type(8)))  _Float16 v8h;
typedef __attribute__((ext_vector_type(8)))  float    v8f;

// ---------------------------------------------------------------------------
// Fragment loaders
//
// 16-bit A-matrix 16x32 layout (ISA 7.12.2):
//   lanes 0-15 : row M=lane,     halves 0-7 -> K 0..7,   halves 8-15 -> K 16..23
//   lanes 16-31: row M=lane-16,  halves 0-7 -> K 8..15,  halves 8-15 -> K 24..31
// So per lane: two contiguous 16B loads at kbase and kbase+16 (kbase = 0 or 8).
//
// B fragments are pre-packed so each lane reads 16 contiguous halves (32B).
// ---------------------------------------------------------------------------
__device__ __forceinline__ v16h load_a_frag(const _Float16* __restrict__ A,
                                            int pitch, int lane, int kt) {
  const int m  = lane & 15;
  const int kb = kt * 32 + ((lane & 16) ? 8 : 0);
  v8h lo = *(const v8h*)(A + (size_t)m * pitch + kb);
  v8h hi = *(const v8h*)(A + (size_t)m * pitch + kb + 16);
  return __builtin_shufflevector(lo, hi, 0, 1, 2, 3, 4, 5, 6, 7,
                                 8, 9, 10, 11, 12, 13, 14, 15);
}

__device__ __forceinline__ v16h load_b_frag(const _Float16* __restrict__ pw,
                                            int ntn, int kt, int nt, int lane) {
  return *(const v16h*)(pw + (((size_t)kt * ntn + nt) << 9) + lane * 16);
}

__device__ __forceinline__ v8f wmma_f16(v16h a, v16h b, v8f c) {
  return __builtin_amdgcn_wmma_f32_16x16x32_f16(false, a, false, b,
                                                (short)0, c, false, false);
}

// ---------------------------------------------------------------------------
// Weight packing: f32 row-major W[din x dout] -> f16 fragment-packed,
// zero padded to Kpad x Npad.  Within each 32x16 (KxN) block:
//   dst[block*512 + lane*16 + (klocal & 7) + ((klocal>>4)<<3)],  lane = (((klocal>>3)&1)?16:0) + n
// ---------------------------------------------------------------------------
__global__ void pack_w_kernel(const float* __restrict__ W, _Float16* __restrict__ dst,
                              int din, int dout, int Kpad, int Npad) {
  int idx = blockIdx.x * blockDim.x + threadIdx.x;
  int total = Kpad * Npad;
  if (idx >= total) return;
  int k = idx / Npad, n = idx % Npad;
  float v = (k < din && n < dout) ? W[(size_t)k * dout + n] : 0.0f;
  int kt = k >> 5, nt = n >> 4;
  int kl = k & 31, nl = n & 15;
  int lane = (((kl >> 3) & 1) ? 16 : 0) + nl;
  int ntn = Npad >> 4;
  dst[(((size_t)kt * ntn + nt) << 9) + lane * 16 + (kl & 7) + ((kl >> 4) << 3)] = (_Float16)v;
}

// ---------------------------------------------------------------------------
// One MLP layer for a single wave's 16-row tile held in LDS.
// in : 16 x Kpad (f16, row-major, pitch Kpad)
// out: 16 x Npad (f16, row-major, pitch Npad), ReLU(acc + bias), pad cols -> 0
// ---------------------------------------------------------------------------
__device__ __forceinline__ void wave_layer(const _Float16* in, int Kpad,
                                           const _Float16* __restrict__ pw,
                                           const float* __restrict__ bias,
                                           int dout, int Npad,
                                           _Float16* out, int lane) {
  const int ktn = Kpad >> 5, ntn = Npad >> 4;
  const int nl = lane & 15;
  const int mbase = (lane & 16) ? 8 : 0;
  for (int nt = 0; nt < ntn; ++nt) {
    v8f acc = {};
    for (int kt = 0; kt < ktn; ++kt) {
      v16h a = load_a_frag(in, Kpad, lane, kt);
      v16h b = load_b_frag(pw, ntn, kt, nt, lane);
      acc = wmma_f16(a, b, acc);
    }
    int n = nt * 16 + nl;
    float bv = (n < dout) ? bias[n] : 0.0f;
#pragma unroll
    for (int r = 0; r < 8; ++r) {
      float v = acc[r] + bv;
      v = v > 0.0f ? v : 0.0f;                 // ReLU
      out[(size_t)(mbase + r) * Npad + n] = (_Float16)v;
    }
  }
}

// ---------------------------------------------------------------------------
// Fused edge-attention kernel: gather + 6-layer MLP + sigmoid, one 16-edge
// tile per wave, all activations in per-wave LDS ping-pong buffers.
// Padded dims: 192 ->128 ->128 ->64 ->32 ->32 ->16(dout=1)
// ---------------------------------------------------------------------------
__global__ void __launch_bounds__(128)
edge_mlp_kernel(const float* __restrict__ jet, const float* __restrict__ nf,
                const float* __restrict__ h,
                const int* __restrict__ esrc, const int* __restrict__ edst,
                const _Float16* __restrict__ pw0, const _Float16* __restrict__ pw1,
                const _Float16* __restrict__ pw2, const _Float16* __restrict__ pw3,
                const _Float16* __restrict__ pw4, const _Float16* __restrict__ pw5,
                const float* __restrict__ b0, const float* __restrict__ b1,
                const float* __restrict__ b2, const float* __restrict__ b3,
                const float* __restrict__ b4, const float* __restrict__ b5,
                float* __restrict__ ew_out, int n_tiles) {
  __shared__ _Float16 smem[4][2][16 * 192];     // 48 KB static LDS
  const int wid  = threadIdx.x >> 5;
  const int lane = threadIdx.x & 31;
  const int tile = blockIdx.x * 4 + wid;
  if (tile >= n_tiles) return;                  // wave-uniform guard

  _Float16* bufA = smem[wid][0];
  _Float16* bufB = smem[wid][1];
  const int e0 = tile * 16;

  // ---- gather z_att = [jet_s(8) | nf_s(16) | h_s(64) | nf_d(16) | h_d(64)] pad->192
  for (int r = 0; r < 16; ++r) {
    const int e = e0 + r;
    const int s = esrc[e], d = edst[e];
    for (int c = lane; c < 192; c += 32) {
      float v;
      if      (c < 8)   v = jet[(size_t)s * 8  + c];
      else if (c < 24)  v = nf [(size_t)s * 16 + (c - 8)];
      else if (c < 88)  v = h  [(size_t)s * 64 + (c - 24)];
      else if (c < 104) v = nf [(size_t)d * 16 + (c - 88)];
      else if (c < 168) v = h  [(size_t)d * 64 + (c - 104)];
      else              v = 0.0f;
      bufA[(size_t)r * 192 + c] = (_Float16)v;
    }
  }

  // ---- MLP chain (all WMMA, f32 accumulate)
  wave_layer(bufA, 192, pw0, b0, 100, 128, bufB, lane);
  wave_layer(bufB, 128, pw1, b1, 100, 128, bufA, lane);
  wave_layer(bufA, 128, pw2, b2,  50,  64, bufB, lane);
  wave_layer(bufB,  64, pw3, b3,  20,  32, bufA, lane);
  wave_layer(bufA,  32, pw4, b4,  10,  32, bufB, lane);

  // ---- final layer (dout = 1) + sigmoid -> e_weight
  {
    v8f acc = {};
    v16h a = load_a_frag(bufB, 32, lane, 0);
    v16h b = load_b_frag(pw5, 1, 0, 0, lane);
    acc = wmma_f16(a, b, acc);
    if ((lane & 15) == 0) {                     // column n == 0 only
      const float bv = b5[0];
      const int mbase = (lane & 16) ? 8 : 0;
#pragma unroll
      for (int r = 0; r < 8; ++r) {
        float v = acc[r] + bv;
        ew_out[e0 + mbase + r] = 1.0f / (1.0f + expf(-v));
      }
    }
  }
}

// ---------------------------------------------------------------------------
// Zero-fill
// ---------------------------------------------------------------------------
__global__ void zero_kernel(float* __restrict__ p, int n) {
  int t = blockIdx.x * blockDim.x + threadIdx.x;
  if (t < n) p[t] = 0.0f;
}

// ---------------------------------------------------------------------------
// Weighted message scatter: agg[dst, :80] += w(e) * [nf_s(16) | h_s(64)]
// ---------------------------------------------------------------------------
__global__ void scatter_msg_kernel(const int* __restrict__ esrc,
                                   const int* __restrict__ edst,
                                   const float* __restrict__ nf,
                                   const float* __restrict__ h,
                                   const float* __restrict__ ew,
                                   float* __restrict__ agg, int total) {
  int t = blockIdx.x * blockDim.x + threadIdx.x;
  if (t >= total) return;
  int e = t / 80, c = t % 80;
  float w = ew[e];
  int s = esrc[e];
  float v = (c < 16) ? nf[(size_t)s * 16 + c] : h[(size_t)s * 64 + (c - 16)];
  atomicAdd(&agg[(size_t)edst[e] * 80 + c], w * v);
}

// ---------------------------------------------------------------------------
// Node-input gather: [nf(16) | h(64) | agg(80) | jet(8)] pad -> 192, f16
// ---------------------------------------------------------------------------
__global__ void node_gather_kernel(const float* __restrict__ nf,
                                   const float* __restrict__ h,
                                   const float* __restrict__ agg,
                                   const float* __restrict__ jet,
                                   _Float16* __restrict__ out, int total) {
  int t = blockIdx.x * blockDim.x + threadIdx.x;
  if (t >= total) return;
  int row = t / 192, c = t % 192;
  float v;
  if      (c < 16)  v = nf [(size_t)row * 16 + c];
  else if (c < 80)  v = h  [(size_t)row * 64 + (c - 16)];
  else if (c < 160) v = agg[(size_t)row * 80 + (c - 80)];
  else if (c < 168) v = jet[(size_t)row * 8  + (c - 160)];
  else              v = 0.0f;
  out[t] = (_Float16)v;
}

// ---------------------------------------------------------------------------
// Node GEMM, template-specialized per layer, 4-way N register blocking:
// each wave computes a 16x64 output tile; the A fragment is loaded once per
// K-step and reused by 4 WMMAs (4x arithmetic intensity on the A stream).
// All loop trip counts are compile-time -> full unroll / pipelining.
// ---------------------------------------------------------------------------
template <int KPAD, int NPAD, bool RELU, bool FINAL>
__global__ void __launch_bounds__(128)
gemm_tile4_kernel(const _Float16* __restrict__ A,
                  const _Float16* __restrict__ pw,
                  const float* __restrict__ bias, int dout,
                  _Float16* __restrict__ outh, float* __restrict__ outf,
                  int Mtiles) {
  constexpr int NTN = NPAD / 16;
  constexpr int KTN = KPAD / 32;
  constexpr int NG  = NPAD / 64;                // groups of 4 n-tiles
  const int lane = threadIdx.x & 31;
  const int tile = blockIdx.x * 4 + (threadIdx.x >> 5);
  if (tile >= Mtiles * NG) return;              // wave-uniform guard
  const int mt = tile / NG;
  const int ng = tile % NG;
  const _Float16* Arow = A + (size_t)mt * 16 * KPAD;

  v8f acc[4] = {v8f{}, v8f{}, v8f{}, v8f{}};
  for (int kt = 0; kt < KTN; ++kt) {
    v16h a = load_a_frag(Arow, KPAD, lane, kt);
#pragma unroll
    for (int j = 0; j < 4; ++j) {
      v16h b = load_b_frag(pw, NTN, kt, ng * 4 + j, lane);
      acc[j] = wmma_f16(a, b, acc[j]);
    }
  }

  const int nl = lane & 15;
  const int mbase = (lane & 16) ? 8 : 0;
#pragma unroll
  for (int j = 0; j < 4; ++j) {
    const int n = (ng * 4 + j) * 16 + nl;
    const float bv = (bias != nullptr && n < dout) ? bias[n] : 0.0f;
#pragma unroll
    for (int r = 0; r < 8; ++r) {
      float v = acc[j][r] + bv;
      if (RELU) v = v > 0.0f ? v : 0.0f;
      const int m = mt * 16 + mbase + r;
      if (FINAL) {
        if (n < dout) outf[(size_t)m * dout + n] = v;
      } else {
        outh[(size_t)m * NPAD + n] = (_Float16)v;
      }
    }
  }
}

// ---------------------------------------------------------------------------
// Host-side orchestration
// ---------------------------------------------------------------------------
extern "C" void kernel_launch(void* const* d_in, const int* in_sizes, int n_in,
                              void* d_out, int out_size, void* d_ws, size_t ws_size,
                              hipStream_t stream) {
  const float* nf   = (const float*)d_in[0];
  const float* h    = (const float*)d_in[1];
  const float* jet  = (const float*)d_in[2];
  const int*   esrc = (const int*)d_in[3];
  const int*   edst = (const int*)d_in[4];

  const int E = in_sizes[3];
  const int N = in_sizes[0] / 16;               // F_N = 16

  // Edge params: (W,b) x 6 starting at index 5
  const float* eW[6]; const float* eB[6];
  for (int i = 0; i < 6; ++i) { eW[i] = (const float*)d_in[5 + 2 * i];
                                eB[i] = (const float*)d_in[6 + 2 * i]; }
  // Node params: first layer has no bias (None is dropped from the pytree)
  const float* nW[5]; const float* nB[5];
  nW[0] = (const float*)d_in[17]; nB[0] = nullptr;
  for (int i = 1; i < 5; ++i) { nW[i] = (const float*)d_in[16 + 2 * i];
                                nB[i] = (const float*)d_in[17 + 2 * i]; }

  // Padded layer geometry: {din, dout, Kpad, Npad}
  // Node Npads are multiples of 64 for uniform 4-tile N blocking; each Npad is
  // also a multiple of 32 so it chains as the next layer's Kpad.
  static const int eg[6][4] = {{168,100,192,128},{100,100,128,128},{100,50,128,64},
                               {50,20,64,32},{20,10,32,32},{10,1,32,16}};
  static const int ng[5][4] = {{168,500,192,512},{500,400,512,448},{400,400,448,448},
                               {400,200,448,256},{200,64,256,64}};

  // Workspace bump allocator (256B aligned)
  char* ws = (char*)d_ws; size_t off = 0;
  auto alloc = [&](size_t bytes) -> char* {
    char* p = ws + off; off = (off + bytes + 255) & ~(size_t)255; return p;
  };
  _Float16* epw[6]; _Float16* npw[5];
  for (int i = 0; i < 6; ++i) epw[i] = (_Float16*)alloc((size_t)eg[i][2] * eg[i][3] * 2);
  for (int i = 0; i < 5; ++i) npw[i] = (_Float16*)alloc((size_t)ng[i][2] * ng[i][3] * 2);
  float*     agg  = (float*)alloc((size_t)N * 80 * 4);
  _Float16*  act0 = (_Float16*)alloc((size_t)N * 192 * 2);
  _Float16*  act1 = (_Float16*)alloc((size_t)N * 512 * 2);
  _Float16*  act2 = (_Float16*)alloc((size_t)N * 448 * 2);
  _Float16*  act3 = (_Float16*)alloc((size_t)N * 448 * 2);
  _Float16*  act4 = (_Float16*)alloc((size_t)N * 256 * 2);

  float* new_h = (float*)d_out;                  // [N x 64]
  float* ew    = (float*)d_out + (size_t)N * 64; // [E]

  // 1) pack all weights into f16 fragment layout (stays L2-resident)
  for (int i = 0; i < 6; ++i) {
    int tot = eg[i][2] * eg[i][3];
    pack_w_kernel<<<(tot + 255) / 256, 256, 0, stream>>>(
        eW[i], epw[i], eg[i][0], eg[i][1], eg[i][2], eg[i][3]);
  }
  for (int i = 0; i < 5; ++i) {
    int tot = ng[i][2] * ng[i][3];
    pack_w_kernel<<<(tot + 255) / 256, 256, 0, stream>>>(
        nW[i], npw[i], ng[i][0], ng[i][1], ng[i][2], ng[i][3]);
  }

  // 2) zero segment-sum accumulator
  {
    int tot = N * 80;
    zero_kernel<<<(tot + 255) / 256, 256, 0, stream>>>(agg, tot);
  }

  // 3) fused edge-attention MLP -> e_weight (16 edges per wave, all-WMMA)
  {
    int tiles = E / 16;
    edge_mlp_kernel<<<(tiles + 3) / 4, 128, 0, stream>>>(
        jet, nf, h, esrc, edst,
        epw[0], epw[1], epw[2], epw[3], epw[4], epw[5],
        eB[0], eB[1], eB[2], eB[3], eB[4], eB[5],
        ew, tiles);
  }

  // 4) weighted message scatter into agg
  {
    int tot = E * 80;
    scatter_msg_kernel<<<(tot + 255) / 256, 256, 0, stream>>>(
        esrc, edst, nf, h, ew, agg, tot);
  }

  // 5) node-input gather (f16, padded to 192)
  {
    int tot = N * 192;
    node_gather_kernel<<<(tot + 255) / 256, 256, 0, stream>>>(
        nf, h, agg, jet, act0, tot);
  }

  // 6) node MLP: 5 template-specialized WMMA GEMM layers (16x64 tile / wave)
  const int Mt = N / 16;
  {
    int tiles = Mt * (512 / 64);
    gemm_tile4_kernel<192, 512, true, false><<<(tiles + 3) / 4, 128, 0, stream>>>(
        act0, npw[0], nB[0], 500, act1, nullptr, Mt);
  }
  {
    int tiles = Mt * (448 / 64);
    gemm_tile4_kernel<512, 448, true, false><<<(tiles + 3) / 4, 128, 0, stream>>>(
        act1, npw[1], nB[1], 400, act2, nullptr, Mt);
  }
  {
    int tiles = Mt * (448 / 64);
    gemm_tile4_kernel<448, 448, true, false><<<(tiles + 3) / 4, 128, 0, stream>>>(
        act2, npw[2], nB[2], 400, act3, nullptr, Mt);
  }
  {
    int tiles = Mt * (256 / 64);
    gemm_tile4_kernel<448, 256, true, false><<<(tiles + 3) / 4, 128, 0, stream>>>(
        act3, npw[3], nB[3], 200, act4, nullptr, Mt);
  }
  {
    int tiles = Mt * (64 / 64);
    gemm_tile4_kernel<256, 64, false, true><<<(tiles + 3) / 4, 128, 0, stream>>>(
        act4, npw[4], nB[4], 64, nullptr, new_h, Mt);
  }
}
